// TempModel_58892591563301
// MI455X (gfx1250) — matrix operand order repaired and checked by reading, not compiled, not run on hardware
//
#include <hip/hip_runtime.h>
#include <hip/hip_bf16.h>
#include <cstdint>
#include <cstddef>

// ---------------- constants ----------------
#define BB      4
#define N0      8192
#define NS      32      // nsample (all levels)
#define NCLASS  13
#define BN_EPS  1e-5f

typedef __attribute__((ext_vector_type(16))) _Float16 v16h;
typedef __attribute__((ext_vector_type(8)))  float    v8f;

static inline int cdiv_h(int a, int b) { return (a + b - 1) / b; }
static inline int pad32_h(int x) { return (x + 31) & ~31; }

// K-block permutation so a lane's 16-element A fragment is contiguous:
// within each 32-block store order [0..7][16..23][8..15][24..31]
__device__ __forceinline__ int kperm(int k) {
  int kb = k & 31, base = k & ~31;
  int p = (kb < 8) ? kb : (kb < 16) ? (kb + 16 - 8 + 8) /*kb+16*/ : (kb < 24) ? (kb - 8) : kb;
  // note: kb in [8,16) -> kb+8 ; kb in [16,24) -> kb-8
  return base + ((kb < 8) ? kb : (kb < 16) ? (kb + 8) : (kb < 24) ? (kb - 8) : kb);
  (void)p;
}

// ---------------- kernels ----------------

// data (B,3,N) -> xyz (B,N,3)
__global__ void transpose_in_kernel(const float* __restrict__ data,
                                    float* __restrict__ xyz, int B_, int N_) {
  int t = blockIdx.x * blockDim.x + threadIdx.x;
  if (t >= B_ * N_) return;
  int n = t % N_, b = t / N_;
  for (int c = 0; c < 3; ++c)
    xyz[((size_t)b * N_ + n) * 3 + c] = data[((size_t)b * 3 + c) * N_ + n];
}

// farthest point sampling; one block (256 threads) per batch; dist kept in LDS
__global__ void fps_kernel(const float* __restrict__ xyz, int* __restrict__ fidx,
                           int Np, int npoint) {
  __shared__ float dist[N0];
  __shared__ float rv[256];
  __shared__ int   ri[256];
  __shared__ int   s_far;
  int b = blockIdx.x, tid = threadIdx.x;
  const float* base = xyz + (size_t)b * Np * 3;
  for (int i = tid; i < Np; i += blockDim.x) dist[i] = 1e10f;
  if (tid == 0) s_far = 0;
  __syncthreads();
  for (int it = 0; it < npoint; ++it) {
    int far = s_far;
    if (tid == 0) fidx[b * npoint + it] = far;
    float cx = base[far*3+0], cy = base[far*3+1], cz = base[far*3+2];
    float bv = -1.0f; int bi = 0;
    for (int i = tid; i < Np; i += blockDim.x) {
      float dx = base[i*3+0]-cx, dy = base[i*3+1]-cy, dz = base[i*3+2]-cz;
      float d = dx*dx + dy*dy + dz*dz;
      float nd = fminf(dist[i], d);
      dist[i] = nd;
      if (nd > bv) { bv = nd; bi = i; }
    }
    rv[tid] = bv; ri[tid] = bi;
    __syncthreads();
    for (int o = 128; o > 0; o >>= 1) {
      if (tid < o) {
        if (rv[tid+o] > rv[tid] || (rv[tid+o] == rv[tid] && ri[tid+o] < ri[tid])) {
          rv[tid] = rv[tid+o]; ri[tid] = ri[tid+o];
        }
      }
      __syncthreads();
    }
    if (tid == 0) s_far = ri[0];
    __syncthreads();
  }
}

__global__ void gather_xyz_kernel(const float* __restrict__ xyz, const int* __restrict__ fidx,
                                  float* __restrict__ nxyz, int B_, int Np, int S) {
  int t = blockIdx.x * blockDim.x + threadIdx.x;
  if (t >= B_ * S) return;
  int s = t % S, b = t / S;
  int p = fidx[b * S + s];
  for (int c = 0; c < 3; ++c)
    nxyz[((size_t)b * S + s) * 3 + c] = xyz[((size_t)b * Np + p) * 3 + c];
}

// first `ns` indices (ascending) with d<=r^2; pad with first found (ref semantics)
__global__ void ball_query_kernel(const float* __restrict__ xyz, const float* __restrict__ nxyz,
                                  int* __restrict__ bidx, int B_, int Np, int S, int ns, float r2) {
  int t = blockIdx.x * blockDim.x + threadIdx.x;
  if (t >= B_ * S) return;
  int s = t % S, b = t / S;
  float cx = nxyz[((size_t)b*S+s)*3+0];
  float cy = nxyz[((size_t)b*S+s)*3+1];
  float cz = nxyz[((size_t)b*S+s)*3+2];
  const float* base = xyz + (size_t)b * Np * 3;
  int* out = bidx + (size_t)t * ns;
  int cnt = 0, first = 0;
  for (int i = 0; i < Np && cnt < ns; ++i) {
    float dx = base[i*3+0]-cx, dy = base[i*3+1]-cy, dz = base[i*3+2]-cz;
    float d = dx*dx + dy*dy + dz*dz;
    if (d <= r2) { if (cnt == 0) first = i; out[cnt++] = i; }
  }
  for (; cnt < ns; ++cnt) out[cnt] = first;
}

// build packed f16 A rows: [xyz(idx)-center (3), pts(idx) (C), 0-pad] with kperm layout
__global__ void group_pack_kernel(const float* __restrict__ xyz, const float* __restrict__ nxyz,
                                  const int* __restrict__ bidx, const float* __restrict__ pts,
                                  _Float16* __restrict__ A, int B_, int Np, int S, int ns,
                                  int C, int Kp) {
  int t = blockIdx.x * blockDim.x + threadIdx.x;
  if (t >= B_ * S * ns) return;
  int k0 = t % ns, s = (t / ns) % S, b = t / (ns * S);
  int pi = bidx[((size_t)b * S + s) * ns + k0];
  const float* p = xyz  + ((size_t)b * Np + pi) * 3;
  const float* c = nxyz + ((size_t)b * S  + s ) * 3;
  const float* pf = pts + ((size_t)b * Np + pi) * C;
  _Float16* row = A + (size_t)t * Kp;
  int cin = 3 + C;
  for (int k = 0; k < Kp; ++k) {
    float v = (k < 3) ? (p[k] - c[k]) : (k < cin) ? pf[k - 3] : 0.0f;
    row[kperm(k)] = (_Float16)v;
  }
}

// pack weights: W (K x Nc, f32) -> Wt (Ncp32 x Kp, f16), transposed + zero-padded
__global__ void pack_wt_kernel(const float* __restrict__ W, _Float16* __restrict__ Wt,
                               int K, int Nc, int Kp, int Ncp) {
  int t = blockIdx.x * blockDim.x + threadIdx.x;
  if (t >= Ncp * Kp) return;
  int k = t % Kp, n = t / Kp;
  float v = (n < Nc && k < K) ? W[(size_t)k * Nc + n] : 0.0f;
  Wt[t] = (_Float16)v;
}

// C[M,Nc] = A[M,Kp](f16 packed) @ Wt^T + bias ; 4 waves/block, 16x32 tile/wave,
// 2 WMMAs per K-step with A-fragment reuse; no branches in the K loop.
__global__ void gemm_wmma_packed(const _Float16* __restrict__ A, const _Float16* __restrict__ Wt,
                                 const float* __restrict__ bias, float* __restrict__ C,
                                 int M, int Kp, int Nc) {
  int wave = threadIdx.x >> 5;
  int lane = threadIdx.x & 31;
  int half = lane >> 4, l16 = lane & 15;
  int tileM = blockIdx.x * 4 + wave;
  if (tileM * 16 >= M) return;                 // wave-uniform
  int row  = tileM * 16 + l16;
  int col0 = blockIdx.y * 32 + l16;

  const _Float16* Ab = A  + (size_t)row * Kp;
  const _Float16* B0 = Wt + (size_t)col0 * Kp;
  const _Float16* B1 = Wt + (size_t)(col0 + 16) * Kp;

  v8f acc0 = {}, acc1 = {};
#pragma unroll 2
  for (int k0 = 0; k0 < Kp; k0 += 32) {
    v16h a  = *(const v16h*)(Ab + k0 + half * 16);
    v16h b0, b1;
    {
      typedef __attribute__((ext_vector_type(8))) _Float16 v8h_;
      v8h_ b0l = *(const v8h_*)(B0 + k0 + half * 8), b0h = *(const v8h_*)(B0 + k0 + 16 + half * 8);
      v8h_ b1l = *(const v8h_*)(B1 + k0 + half * 8), b1h = *(const v8h_*)(B1 + k0 + 16 + half * 8);
      b0 = __builtin_shufflevector(b0l, b0h, 0,1,2,3,4,5,6,7,8,9,10,11,12,13,14,15);
      b1 = __builtin_shufflevector(b1l, b1h, 0,1,2,3,4,5,6,7,8,9,10,11,12,13,14,15);
    }
    acc0 = __builtin_amdgcn_wmma_f32_16x16x32_f16(false, a, false, b0, (short)0, acc0, false, false);
    acc1 = __builtin_amdgcn_wmma_f32_16x16x32_f16(false, a, false, b1, (short)0, acc1, false, false);
  }
  // D layout: vgpr r -> row = 16*tileM + half*8 + r ; col = col0 / col0+16
#pragma unroll
  for (int r = 0; r < 8; ++r) {
    int rr = tileM * 16 + half * 8 + r;        // rr < M (M is a multiple of 16)
    if (col0 < Nc)      C[(size_t)rr * Nc + col0]      = acc0[r] + bias[col0];
    if (col0 + 16 < Nc) C[(size_t)rr * Nc + col0 + 16] = acc1[r] + bias[col0 + 16];
  }
}

// per-channel mean / rstd over M rows (population var, matches jnp.var)
__global__ void bn_stats_kernel(const float* __restrict__ X, float* __restrict__ mean,
                                float* __restrict__ rstd, int M, int C) {
  __shared__ float rs[256], rs2[256];
  int c = blockIdx.x, tid = threadIdx.x;
  float s = 0.f, s2 = 0.f;
  for (int i = tid; i < M; i += blockDim.x) {
    float v = X[(size_t)i * C + c];
    s += v; s2 += v * v;
  }
  rs[tid] = s; rs2[tid] = s2;
  __syncthreads();
  for (int o = 128; o > 0; o >>= 1) {
    if (tid < o) { rs[tid] += rs[tid + o]; rs2[tid] += rs2[tid + o]; }
    __syncthreads();
  }
  if (tid == 0) {
    float m = rs[0] / (float)M;
    float v = fmaxf(rs2[0] / (float)M - m * m, 0.0f);
    mean[c] = m;
    rstd[c] = rsqrtf(v + BN_EPS);
  }
}

// BN+ReLU in-place on f32, optionally also emit packed f16 (kperm layout) for next GEMM
__global__ void bn_relu_pack_kernel(float* __restrict__ X, const float* __restrict__ mean,
                                    const float* __restrict__ rstd, const float* __restrict__ g,
                                    const float* __restrict__ bt, _Float16* __restrict__ P,
                                    int M, int C, int Kp) {
  int t = blockIdx.x * blockDim.x + threadIdx.x;
  if (t >= M * Kp) return;
  int k = t % Kp, i = t / Kp;
  if (k < C) {
    float y = fmaxf((X[(size_t)i * C + k] - mean[k]) * rstd[k] * g[k] + bt[k], 0.0f);
    X[(size_t)i * C + k] = y;
    if (P) P[(size_t)i * Kp + kperm(k)] = (_Float16)y;
  } else if (P) {
    P[(size_t)i * Kp + kperm(k)] = (_Float16)0.0f;
  }
}

__global__ void maxpool_kernel(const float* __restrict__ X, float* __restrict__ out,
                               int BS, int ns, int C) {
  int t = blockIdx.x * blockDim.x + threadIdx.x;
  if (t >= BS * C) return;
  int c = t % C, q = t / C;
  float m = -1e30f;
  for (int k = 0; k < ns; ++k)
    m = fmaxf(m, X[((size_t)q * ns + k) * C + c]);
  out[(size_t)q * C + c] = m;
}

// 3 nearest neighbors of xyz1 points among xyz2 points + inverse-dist weights
__global__ void knn3_kernel(const float* __restrict__ xyz1, const float* __restrict__ xyz2,
                            int* __restrict__ idx3, float* __restrict__ w3,
                            int B_, int N1, int N2) {
  int t = blockIdx.x * blockDim.x + threadIdx.x;
  if (t >= B_ * N1) return;
  int n = t % N1, b = t / N1;
  float px = xyz1[((size_t)b*N1+n)*3+0];
  float py = xyz1[((size_t)b*N1+n)*3+1];
  float pz = xyz1[((size_t)b*N1+n)*3+2];
  const float* base = xyz2 + (size_t)b * N2 * 3;
  float d0 = 1e30f, d1 = 1e30f, d2 = 1e30f;
  int   i0 = 0, i1 = 0, i2 = 0;
  for (int j = 0; j < N2; ++j) {
    float dx = base[j*3+0]-px, dy = base[j*3+1]-py, dz = base[j*3+2]-pz;
    float d = dx*dx + dy*dy + dz*dz;
    if (d < d0)      { d2=d1; i2=i1; d1=d0; i1=i0; d0=d; i0=j; }
    else if (d < d1) { d2=d1; i2=i1; d1=d;  i1=j; }
    else if (d < d2) { d2=d;  i2=j; }
  }
  float w0 = 1.0f / (fmaxf(d0, 0.f) + 1e-8f);
  float w1 = 1.0f / (fmaxf(d1, 0.f) + 1e-8f);
  float w2 = 1.0f / (fmaxf(d2, 0.f) + 1e-8f);
  float ws = w0 + w1 + w2;
  int* id = idx3 + (size_t)t * 3;
  float* w = w3 + (size_t)t * 3;
  id[0]=i0; id[1]=i1; id[2]=i2;
  w[0]=w0/ws; w[1]=w1/ws; w[2]=w2/ws;
}

// packed f16 A[(b,n),:] = [points1 (C1), sum_j w_j*points2[idx_j] (C2), 0-pad], kperm layout
__global__ void interp_pack_kernel(const float* __restrict__ p1, const float* __restrict__ p2,
                                   const int* __restrict__ idx3, const float* __restrict__ w3,
                                   _Float16* __restrict__ A, int B_, int N1, int N2,
                                   int C1, int C2, int Kp) {
  int t = blockIdx.x * blockDim.x + threadIdx.x;
  if (t >= B_ * N1 * Kp) return;
  int k = t % Kp, n = (t / Kp) % N1, b = t / (Kp * N1);
  size_t q = (size_t)b * N1 + n;
  float v = 0.0f;
  if (k < C1) {
    v = p1[q * C1 + k];
  } else if (k < C1 + C2) {
    int cc = k - C1;
    const int*   id = idx3 + q * 3;
    const float* w  = w3   + q * 3;
    const float* pb = p2 + (size_t)b * N2 * C2;
    v = w[0] * pb[(size_t)id[0] * C2 + cc]
      + w[1] * pb[(size_t)id[1] * C2 + cc]
      + w[2] * pb[(size_t)id[2] * C2 + cc];
  }
  A[q * Kp + kperm(k)] = (_Float16)v;
}

// log_softmax over last dim (13) + transpose (B,N,13)->(B,13,N)
__global__ void logsoftmax_out_kernel(const float* __restrict__ logits, float* __restrict__ out,
                                      int B_, int N_, int C) {
  int t = blockIdx.x * blockDim.x + threadIdx.x;
  if (t >= B_ * N_) return;
  int n = t % N_, b = t / N_;
  const float* L = logits + (size_t)t * C;
  float m = -1e30f;
  for (int c = 0; c < C; ++c) m = fmaxf(m, L[c]);
  float ssum = 0.f;
  for (int c = 0; c < C; ++c) ssum += expf(L[c] - m);
  float lse = m + logf(ssum);
  for (int c = 0; c < C; ++c)
    out[((size_t)b * C + c) * N_ + n] = L[c] - lse;
}

// ---------------- host orchestration ----------------

struct LayerP {
  const float* W; const float* b; const float* g; const float* bt;
  int cin, cout, Kp, Ncp;
  _Float16* Wt;
};

static void run_layers(const _Float16* inPack, _Float16* pA, _Float16* pB, float* scratchF,
                       float* outLast, _Float16* lastPack, const LayerP* L, int nl, int M,
                       float* mean, float* rstd, hipStream_t s) {
  const _Float16* src = inPack;
  for (int li = 0; li < nl; ++li) {
    float* dst = (li == nl - 1) ? outLast : scratchF;
    dim3 grid(cdiv_h(M / 16, 4), L[li].Ncp / 32);
    gemm_wmma_packed<<<grid, 128, 0, s>>>(src, L[li].Wt, L[li].b, dst, M, L[li].Kp, L[li].cout);
    bn_stats_kernel<<<L[li].cout, 256, 0, s>>>(dst, mean, rstd, M, L[li].cout);
    _Float16* nxt = (li == nl - 1) ? lastPack : ((src == pA) ? pB : pA);
    int KpO = pad32_h(L[li].cout);
    int tot = M * KpO;
    bn_relu_pack_kernel<<<cdiv_h(tot, 256), 256, 0, s>>>(dst, mean, rstd, L[li].g, L[li].bt,
                                                         nxt, M, L[li].cout, KpO);
    src = nxt;
  }
}

extern "C" void kernel_launch(void* const* d_in, const int* in_sizes, int n_in,
                              void* d_out, int out_size, void* d_ws, size_t ws_size,
                              hipStream_t stream) {
  (void)in_sizes; (void)n_in; (void)out_size; (void)ws_size;
  auto F = [&](int i) { return (const float*)d_in[i]; };

  // ---- parse inputs (JAX pytree order: sorted dict keys: data, params{fp,head{W2,b2,c1},sa}) ----
  int cur = 0;
  const float* data = F(cur++);

  const int fp_nl[4]       = {2, 2, 2, 3};
  const int fp_cin[4]      = {768, 384, 320, 128};
  const int fp_chans[4][3] = {{256,256,0},{256,256,0},{256,128,0},{128,128,128}};
  LayerP fpL[4][3];
  for (int m = 0; m < 4; ++m) {
    int c = fp_cin[m];
    for (int l = 0; l < fp_nl[m]; ++l) {
      fpL[m][l] = { F(cur), F(cur+1), F(cur+2), F(cur+3), c, fp_chans[m][l],
                    pad32_h(c), pad32_h(fp_chans[m][l]) < 32 ? 32 : pad32_h(fp_chans[m][l]), nullptr };
      cur += 4; c = fp_chans[m][l];
    }
  }
  const float* headW2 = F(cur++);
  const float* headb2 = F(cur++);
  LayerP headc1 = { F(cur), F(cur+1), F(cur+2), F(cur+3), 128, 128, 128, 128, nullptr }; cur += 4;

  const int sa_cin[4]      = {6, 67, 131, 259};
  const int sa_chans[4][3] = {{32,32,64},{64,64,128},{128,128,256},{256,256,512}};
  LayerP saL[4][3];
  for (int m = 0; m < 4; ++m) {
    int c = sa_cin[m];
    for (int l = 0; l < 3; ++l) {
      saL[m][l] = { F(cur), F(cur+1), F(cur+2), F(cur+3), c, sa_chans[m][l],
                    pad32_h(c), pad32_h(sa_chans[m][l]), nullptr };
      cur += 4; c = sa_chans[m][l];
    }
  }

  // ---- workspace bump allocator ----
  size_t off = 0;
  auto alloc = [&](size_t nbytes) -> void* {
    size_t p = (off + 255) & ~(size_t)255;
    off = p + nbytes;
    return (void*)((char*)d_ws + p);
  };

  const int NpLev[5] = {8192, 1024, 256, 64, 16};
  const int Cpts[5]  = {3, 64, 128, 256, 512};

  float* xyzL[5];
  for (int i = 0; i < 5; ++i) xyzL[i] = (float*)alloc((size_t)BB * NpLev[i] * 3 * sizeof(float));
  float* ptsL[5];
  ptsL[0] = xyzL[0];   // l0_points == l0_xyz (data has 3 channels)
  for (int i = 1; i < 5; ++i) ptsL[i] = (float*)alloc((size_t)BB * NpLev[i] * Cpts[i] * sizeof(float));

  int*   fidx = (int*)alloc((size_t)BB * 1024 * sizeof(int));
  int*   bidx = (int*)alloc((size_t)BB * 1024 * NS * sizeof(int));
  size_t bigN = (size_t)BB * 1024 * NS * 64;                 // worst-case M*C
  _Float16* packA = (_Float16*)alloc(bigN * sizeof(_Float16));
  _Float16* packB = (_Float16*)alloc(bigN * sizeof(_Float16));
  float* bufF = (float*)alloc(bigN * sizeof(float));         // f32 GEMM output scratch
  float* mean = (float*)alloc(512 * sizeof(float));
  float* rstd = (float*)alloc(512 * sizeof(float));
  int*   idx3 = (int*)alloc((size_t)BB * N0 * 3 * sizeof(int));
  float* w3   = (float*)alloc((size_t)BB * N0 * 3 * sizeof(float));
  float* fpo[4];
  const int fpoN[4] = {64, 256, 1024, 8192};
  const int fpoC[4] = {256, 256, 128, 128};
  for (int j = 0; j < 4; ++j) fpo[j] = (float*)alloc((size_t)BB * fpoN[j] * fpoC[j] * sizeof(float));
  float* logits = (float*)alloc((size_t)BB * N0 * NCLASS * sizeof(float));

  // ---- pack all weights (f32 KxN -> f16 transposed, padded) ----
  LayerP* allL[22]; int nAll = 0;
  for (int m = 0; m < 4; ++m) for (int l = 0; l < fp_nl[m]; ++l) allL[nAll++] = &fpL[m][l];
  allL[nAll++] = &headc1;
  for (int m = 0; m < 4; ++m) for (int l = 0; l < 3; ++l) allL[nAll++] = &saL[m][l];
  for (int i = 0; i < nAll; ++i) {
    LayerP* L = allL[i];
    L->Wt = (_Float16*)alloc((size_t)L->Ncp * L->Kp * sizeof(_Float16));
    int tot = L->Ncp * L->Kp;
    pack_wt_kernel<<<cdiv_h(tot, 256), 256, 0, stream>>>(L->W, L->Wt, L->cin, L->cout,
                                                         L->Kp, L->Ncp);
  }
  // final classifier weight (no BN): K=128 -> Nc=13, Ncp=32
  _Float16* W2t = (_Float16*)alloc((size_t)32 * 128 * sizeof(_Float16));
  pack_wt_kernel<<<cdiv_h(32 * 128, 256), 256, 0, stream>>>(headW2, W2t, 128, NCLASS, 128, 32);

  // ---- l0 ----
  transpose_in_kernel<<<cdiv_h(BB * N0, 256), 256, 0, stream>>>(data, xyzL[0], BB, N0);

  // ---- set abstraction levels ----
  const float R2 = 32.0f * 32.0f;
  for (int i = 0; i < 4; ++i) {
    int Np = NpLev[i], S = NpLev[i + 1], C = Cpts[i];
    int cout = sa_chans[i][2];
    fps_kernel<<<BB, 256, 0, stream>>>(xyzL[i], fidx, Np, S);
    gather_xyz_kernel<<<cdiv_h(BB * S, 256), 256, 0, stream>>>(xyzL[i], fidx, xyzL[i + 1], BB, Np, S);
    ball_query_kernel<<<cdiv_h(BB * S, 256), 256, 0, stream>>>(xyzL[i], xyzL[i + 1], bidx,
                                                               BB, Np, S, NS, R2);
    int M = BB * S * NS;
    group_pack_kernel<<<cdiv_h(M, 256), 256, 0, stream>>>(xyzL[i], xyzL[i + 1], bidx, ptsL[i],
                                                          packA, BB, Np, S, NS, C, saL[i][0].Kp);
    run_layers(packA, packA, packB, bufF, bufF, nullptr, saL[i], 3, M, mean, rstd, stream);
    maxpool_kernel<<<cdiv_h(BB * S * cout, 256), 256, 0, stream>>>(bufF, ptsL[i + 1], BB * S, NS, cout);
  }

  // ---- feature propagation levels ----
  for (int j = 0; j < 4; ++j) {
    int fa = 3 - j, fb = 4 - j;
    int N1 = NpLev[fa], N2 = NpLev[fb];
    const float* p1 = (j == 3) ? nullptr : ptsL[fa];
    const float* p2 = (j == 0) ? ptsL[4] : fpo[j - 1];
    int C1 = (j == 3) ? 0 : Cpts[fa];
    int C2 = (j == 0) ? Cpts[4] : fp_chans[j - 1][fp_nl[j - 1] - 1];
    knn3_kernel<<<cdiv_h(BB * N1, 256), 256, 0, stream>>>(xyzL[fa], xyzL[fb], idx3, w3, BB, N1, N2);
    int Kp0 = fpL[j][0].Kp;
    int tot = BB * N1 * Kp0;
    interp_pack_kernel<<<cdiv_h(tot, 256), 256, 0, stream>>>(p1, p2, idx3, w3, packA,
                                                             BB, N1, N2, C1, C2, Kp0);
    // FP4 feeds the head: keep the packed f16 of its last layer in packB
    _Float16* lastPack = (j == 3) ? packB : nullptr;
    run_layers(packA, packA, packB, bufF, fpo[j], lastPack, fpL[j], fp_nl[j], BB * N1,
               mean, rstd, stream);
  }

  // ---- head ----
  {
    int M = BB * N0;
    // c1: 128 -> 128 with BN+ReLU, pack to packA
    dim3 g1(cdiv_h(M / 16, 4), headc1.Ncp / 32);
    gemm_wmma_packed<<<g1, 128, 0, stream>>>(packB, headc1.Wt, headc1.b, bufF, M,
                                             headc1.Kp, headc1.cout);
    bn_stats_kernel<<<128, 256, 0, stream>>>(bufF, mean, rstd, M, 128);
    bn_relu_pack_kernel<<<cdiv_h(M * 128, 256), 256, 0, stream>>>(bufF, mean, rstd,
                                                                  headc1.g, headc1.bt,
                                                                  packA, M, 128, 128);
    // W2: 128 -> 13 (+bias, no BN)
    dim3 g2(cdiv_h(M / 16, 4), 1);
    gemm_wmma_packed<<<g2, 128, 0, stream>>>(packA, W2t, headb2, logits, M, 128, NCLASS);
    logsoftmax_out_kernel<<<cdiv_h(BB * N0, 256), 256, 0, stream>>>(logits, (float*)d_out,
                                                                    BB, N0, NCLASS);
  }
}
